// GuidedGNN_12214886990731
// MI455X (gfx1250) — compile-verified, matrix operand order checked by reading, not run on hardware
//
#include <hip/hip_runtime.h>
#include <hip/hip_bf16.h>

typedef __bf16 bf16;
typedef __attribute__((ext_vector_type(16))) __bf16 bf16x16;
typedef __attribute__((ext_vector_type(8)))  __bf16 bf16x8;
typedef __attribute__((ext_vector_type(8)))  float  f32x8;

static constexpr int NN  = 50000;   // nodes
static constexpr int EE  = 800000;  // edges (before self loops)
static constexpr int HH  = 3;       // heads
static constexpr int HE  = 384;     // H*EMB

// ---------------------------------------------------------------------------
// WMMA GEMM:  C[M,N] = A[M,K] * Bt[N,K]^T (+bias[N]) (+ELU)
// A, Bt are bf16 with K contiguous.  M%16==0, N%64==0, K%32==0.
// One wave -> 16x64 strip (4 wmma tiles), A fragment reused across the strip.
// ---------------------------------------------------------------------------
__device__ inline bf16x16 load_frag16(const bf16* p) {
  bf16x8 lo = *(const bf16x8*)(p);
  bf16x8 hi = *(const bf16x8*)(p + 16);
  bf16x16 o;
#pragma unroll
  for (int i = 0; i < 8; ++i) { o[i] = lo[i]; o[i + 8] = hi[i]; }
  return o;
}

template <int ACT>  // 0 = none, 1 = ELU
__global__ void gemm_wmma_bf16(const bf16* __restrict__ A, const bf16* __restrict__ Bt,
                               const float* __restrict__ bias, float* __restrict__ C,
                               int M, int N, int K) {
  const int wid    = (int)(blockIdx.x * (blockDim.x >> 5)) + (int)(threadIdx.x >> 5);
  const int tilesN = N >> 6;
  const int total  = (M >> 4) * tilesN;
  if (wid >= total) return;                 // whole-wave guard: EXEC stays uniform
  const int m0   = (wid / tilesN) << 4;
  const int n0   = (wid % tilesN) << 6;
  const int lane = threadIdx.x & 31;
  const int half = lane >> 4;               // K-halves per 16-bit A-fragment layout
  const int r    = lane & 15;

  const bf16* ap = A  + (size_t)(m0 + r) * K + half * 8;
  const bf16* bp = Bt + (size_t)(n0 + r) * K + half * 8;
  const size_t bstr = (size_t)16 * K;

  f32x8 acc0 = {}, acc1 = {}, acc2 = {}, acc3 = {};
  for (int k = 0; k < K; k += 32) {
    bf16x16 a  = load_frag16(ap + k);
    bf16x16 b0 = load_frag16(bp + k);
    bf16x16 b1 = load_frag16(bp + bstr + k);
    bf16x16 b2 = load_frag16(bp + 2 * bstr + k);
    bf16x16 b3 = load_frag16(bp + 3 * bstr + k);
    acc0 = __builtin_amdgcn_wmma_f32_16x16x32_bf16(false, a, false, b0, (short)0, acc0, false, false);
    acc1 = __builtin_amdgcn_wmma_f32_16x16x32_bf16(false, a, false, b1, (short)0, acc1, false, false);
    acc2 = __builtin_amdgcn_wmma_f32_16x16x32_bf16(false, a, false, b2, (short)0, acc2, false, false);
    acc3 = __builtin_amdgcn_wmma_f32_16x16x32_bf16(false, a, false, b3, (short)0, acc3, false, false);
  }

  f32x8 accs[4] = {acc0, acc1, acc2, acc3};
#pragma unroll
  for (int t = 0; t < 4; ++t) {
    const int col = n0 + t * 16 + r;
    const float bv = bias ? bias[col] : 0.0f;
#pragma unroll
    for (int i = 0; i < 8; ++i) {
      const int row = m0 + half * 8 + i;    // C/D layout: VGPR i -> M = i (+8 for hi lanes)
      float v = accs[t][i] + bv;
      if (ACT == 1) v = v > 0.0f ? v : (__expf(v) - 1.0f);
      C[(size_t)row * N + col] = v;
    }
  }
}

// ---------------------------------------------------------------------------
// Small helper kernels
// ---------------------------------------------------------------------------
__global__ void k_f32_to_bf16(const float* __restrict__ s, bf16* __restrict__ d, long n) {
  long i = (long)blockIdx.x * blockDim.x + threadIdx.x;
  if (i < n) d[i] = (bf16)s[i];
}

// src f32 [R,C] -> dst bf16 [C,R]
__global__ void k_transpose_bf16(const float* __restrict__ s, bf16* __restrict__ d, int R, int C) {
  long i = (long)blockIdx.x * blockDim.x + threadIdx.x;
  if (i < (long)R * C) {
    int rr = (int)(i / C), cc = (int)(i % C);
    d[(size_t)cc * R + rr] = (bf16)s[i];
  }
}

__global__ void k_fill_f32(float* __restrict__ p, float v, long n) {
  long i = (long)blockIdx.x * blockDim.x + threadIdx.x;
  if (i < n) p[i] = v;
}

// out_bf16 = elu(in + bias)   (GAT bias is applied *after* aggregation, as in ref)
__global__ void k_bias_act_to_bf16(const float* __restrict__ in, const float* __restrict__ b,
                                   bf16* __restrict__ out, long n, int C, int act) {
  long i = (long)blockIdx.x * blockDim.x + threadIdx.x;
  if (i < n) {
    float v = in[i] + (b ? b[i % C] : 0.0f);
    if (act) v = v > 0.0f ? v : (__expf(v) - 1.0f);
    out[i] = (bf16)v;
  }
}

// per-(node,head) attention coefficients; one wave per (node,head)
__global__ void k_attn_coeff(const float* __restrict__ h, const float* __restrict__ a_s,
                             const float* __restrict__ a_d, float* __restrict__ als,
                             float* __restrict__ ald, int n) {
  int gw = (int)(blockIdx.x * (blockDim.x >> 5)) + (int)(threadIdx.x >> 5);
  if (gw >= n * HH) return;
  int lane = threadIdx.x & 31;
  int node = gw / HH, hd = gw % HH;
  const float* hp = h + (size_t)node * HE + hd * 128;
  const float* sp = a_s + hd * 128;
  const float* dp = a_d + hd * 128;
  float ss = 0.f, sd = 0.f;
#pragma unroll
  for (int i = 0; i < 4; ++i) {
    int c = lane + 32 * i;
    float hv = hp[c];
    ss += hv * sp[c];
    sd += hv * dp[c];
  }
#pragma unroll
  for (int off = 16; off; off >>= 1) {
    ss += __shfl_xor(ss, off, 32);
    sd += __shfl_xor(sd, off, 32);
  }
  if (lane == 0) { als[gw] = ss; ald[gw] = sd; }
}

__device__ inline void atomicMaxF(float* addr, float val) {
  if (val >= 0.0f) atomicMax((int*)addr, __float_as_int(val));
  else             atomicMin((unsigned int*)addr, __float_as_uint(val));
}

__device__ inline void edge_src_dst(const int* __restrict__ ei, int e, int& src, int& dst) {
  if (e < EE) { src = ei[e]; dst = ei[EE + e]; }
  else        { src = dst = e - EE; }       // self loop
}

__global__ void k_edge_max(const int* __restrict__ ei, const float* __restrict__ als,
                           const float* __restrict__ ald, float* __restrict__ m, int etot) {
  int idx = blockIdx.x * blockDim.x + threadIdx.x;
  if (idx >= etot * HH) return;
  int e = idx / HH, hd = idx - e * HH;
  int src, dst; edge_src_dst(ei, e, src, dst);
  float v = als[src * HH + hd] + ald[dst * HH + hd];
  v = v > 0.0f ? v : 0.2f * v;              // leaky_relu(0.2)
  atomicMaxF(&m[dst * HH + hd], v);
}

__global__ void k_edge_sum(const int* __restrict__ ei, const float* __restrict__ als,
                           const float* __restrict__ ald, const float* __restrict__ m,
                           float* __restrict__ den, int etot) {
  int idx = blockIdx.x * blockDim.x + threadIdx.x;
  if (idx >= etot * HH) return;
  int e = idx / HH, hd = idx - e * HH;
  int src, dst; edge_src_dst(ei, e, src, dst);
  float v = als[src * HH + hd] + ald[dst * HH + hd];
  v = v > 0.0f ? v : 0.2f * v;
  atomicAdd(&den[dst * HH + hd], __expf(v - m[dst * HH + hd]));
}

// one 96-thread block per edge; thread t handles channels [4t, 4t+3]
__global__ void k_edge_agg(const int* __restrict__ ei, const float* __restrict__ h,
                           const float* __restrict__ als, const float* __restrict__ ald,
                           const float* __restrict__ m, const float* __restrict__ den,
                           float* __restrict__ out, int etot) {
  int e = blockIdx.x;
  if (e >= etot) return;
  int src, dst; edge_src_dst(ei, e, src, dst);
  int c  = threadIdx.x * 4;
  int hd = c >> 7;
  float v = als[src * HH + hd] + ald[dst * HH + hd];
  v = v > 0.0f ? v : 0.2f * v;
  float alpha = __expf(v - m[dst * HH + hd]) / (den[dst * HH + hd] + 1e-16f);
  const float4 hv = *(const float4*)(h + (size_t)src * HE + c);
  float* op = out + (size_t)dst * HE + c;
  atomicAdd(op + 0, alpha * hv.x);
  atomicAdd(op + 1, alpha * hv.y);
  atomicAdd(op + 2, alpha * hv.z);
  atomicAdd(op + 3, alpha * hv.w);
}

// final: sigmoid(h[N,512] @ Wl2[512] + bl2); one wave per node
__global__ void k_mlp2_sigmoid(const float* __restrict__ h, const float* __restrict__ w,
                               const float* __restrict__ b, float* __restrict__ out, int n) {
  int wid = (int)(blockIdx.x * (blockDim.x >> 5)) + (int)(threadIdx.x >> 5);
  if (wid >= n) return;
  int lane = threadIdx.x & 31;
  const float* hp = h + (size_t)wid * 512;
  float s = 0.f;
#pragma unroll
  for (int i = 0; i < 16; ++i) { int c = lane + 32 * i; s += hp[c] * w[c]; }
#pragma unroll
  for (int off = 16; off; off >>= 1) s += __shfl_xor(s, off, 32);
  if (lane == 0) out[wid] = 1.0f / (1.0f + __expf(-(s + b[0])));
}

// ---------------------------------------------------------------------------
extern "C" void kernel_launch(void* const* d_in, const int* in_sizes, int n_in,
                              void* d_out, int out_size, void* d_ws, size_t ws_size,
                              hipStream_t stream) {
  (void)in_sizes; (void)n_in; (void)out_size; (void)ws_size;
  const float* x   = (const float*)d_in[0];
  const int*   ei  = (const int*)d_in[1];            // harness contract: ints -> int32
  const float* W[3]  = {(const float*)d_in[2],  (const float*)d_in[6],  (const float*)d_in[10]};
  const float* AS[3] = {(const float*)d_in[3],  (const float*)d_in[7],  (const float*)d_in[11]};
  const float* AD[3] = {(const float*)d_in[4],  (const float*)d_in[8],  (const float*)d_in[12]};
  const float* B[3]  = {(const float*)d_in[5],  (const float*)d_in[9],  (const float*)d_in[13]};
  const float* Wh[3] = {(const float*)d_in[14], (const float*)d_in[16], (const float*)d_in[18]};
  const float* Bh[3] = {(const float*)d_in[15], (const float*)d_in[17], (const float*)d_in[19]};
  const float* Wl1 = (const float*)d_in[20];
  const float* bl1 = (const float*)d_in[21];
  const float* Wl2 = (const float*)d_in[22];
  const float* bl2 = (const float*)d_in[23];
  float* out = (float*)d_out;

  // ---- workspace carve (aligned 256B) ----
  char* ws = (char*)d_ws;
  size_t off = 0;
  auto carve = [&](size_t bytes) -> void* {
    void* p = ws + off;
    off = (off + bytes + 255) & ~(size_t)255;
    return p;
  };
  float* hfull = (float*)carve((size_t)NN * HE * 4);    // 76.8 MB  pre-activation h
  float* agg   = (float*)carve((size_t)NN * 512 * 4);   // 102.4 MB aggregation / GEMM out
  bf16*  act   = (bf16*) carve((size_t)NN * HE * 2);    // 38.4 MB  bf16 activations
  float* als   = (float*)carve((size_t)NN * HH * 4);
  float* ald   = (float*)carve((size_t)NN * HH * 4);
  float* segm  = (float*)carve((size_t)NN * HH * 4);
  float* den   = (float*)carve((size_t)NN * HH * 4);
  bf16*  Wt[3], *Wht[3];
  for (int l = 0; l < 3; ++l) Wt[l]  = (bf16*)carve((size_t)HE * 128 * 2);  // [384,128]
  for (int l = 0; l < 3; ++l) Wht[l] = (bf16*)carve((size_t)128 * HE * 2);  // [128,384]
  bf16* Wl1t = (bf16*)carve((size_t)512 * 128 * 2);                         // [512,128]

  const int ETOT = EE + NN;
  auto cdiv = [](long a, long b) { return (unsigned)((a + b - 1) / b); };

  // ---- weight prep: f32 -> bf16 transposed [N,K] ----
  for (int l = 0; l < 3; ++l) {
    k_transpose_bf16<<<cdiv(128L * HE, 256), 256, 0, stream>>>(W[l],  Wt[l],  128, HE);
    k_transpose_bf16<<<cdiv((long)HE * 128, 256), 256, 0, stream>>>(Wh[l], Wht[l], HE, 128);
  }
  k_transpose_bf16<<<cdiv(128L * 512, 256), 256, 0, stream>>>(Wl1, Wl1t, 128, 512);

  // ---- layer input: x -> bf16 ----
  k_f32_to_bf16<<<cdiv((long)NN * 128, 256), 256, 0, stream>>>(x, act, (long)NN * 128);

  for (int l = 0; l < 3; ++l) {
    // h = act @ W  (no bias; attention reads raw h)   [N,384]
    {
      unsigned waves = (NN / 16) * (HE / 64);
      gemm_wmma_bf16<0><<<cdiv(waves, 4), 128, 0, stream>>>(act, Wt[l], nullptr, hfull, NN, HE, 128);
    }
    // attention coefficients
    k_attn_coeff<<<cdiv((long)NN * HH, 4), 128, 0, stream>>>(hfull, AS[l], AD[l], als, ald, NN);
    // segment softmax state + aggregation target
    k_fill_f32<<<cdiv((long)NN * HH, 256), 256, 0, stream>>>(segm, -__builtin_inff(), (long)NN * HH);
    k_fill_f32<<<cdiv((long)NN * HH, 256), 256, 0, stream>>>(den, 0.0f, (long)NN * HH);
    k_fill_f32<<<cdiv((long)NN * HE, 256), 256, 0, stream>>>(agg, 0.0f, (long)NN * HE);
    // 3-pass segment softmax + weighted scatter (L2-resident tables)
    k_edge_max<<<cdiv((long)ETOT * HH, 256), 256, 0, stream>>>(ei, als, ald, segm, ETOT);
    k_edge_sum<<<cdiv((long)ETOT * HH, 256), 256, 0, stream>>>(ei, als, ald, segm, den, ETOT);
    k_edge_agg<<<ETOT, 96, 0, stream>>>(ei, hfull, als, ald, segm, den, agg, ETOT);
    // elu(agg + b) -> bf16
    k_bias_act_to_bf16<<<cdiv((long)NN * HE, 256), 256, 0, stream>>>(agg, B[l], act, (long)NN * HE, HE, 1);
    // head transform: [N,384] @ [384,128] + bh (no activation)
    {
      unsigned waves = (NN / 16) * (128 / 64);
      gemm_wmma_bf16<0><<<cdiv(waves, 4), 128, 0, stream>>>(act, Wht[l], Bh[l], agg, NN, 128, HE);
    }
    k_f32_to_bf16<<<cdiv((long)NN * 128, 256), 256, 0, stream>>>(agg, act, (long)NN * 128);
  }

  // MLP1: elu([N,128] @ [128,512] + bl1)  (ELU fused in epilogue)
  {
    unsigned waves = (NN / 16) * (512 / 64);
    gemm_wmma_bf16<1><<<cdiv(waves, 4), 128, 0, stream>>>(act, Wl1t, bl1, agg, NN, 512, 128);
  }
  // MLP2 + sigmoid
  k_mlp2_sigmoid<<<cdiv((long)NN, 4), 128, 0, stream>>>(agg, Wl2, bl2, out, NN);
}
